// Network_84026740179249
// MI455X (gfx1250) — matrix operand order repaired and checked by reading, not compiled, not run on hardware
//
#include <hip/hip_runtime.h>
#include <hip/hip_bf16.h>

// ---------------- problem constants ----------------
#define L_STEPS 24
#define NROWS   16384
#define HDIM    128
#define SQ      32
#define D_OP    16
#define D_ATTR  56
#define D_FILT  74
#define D_OUT   32
#define D_X     136      // op(16)+attr(56)+fd(32)+od(32)
#define KPAD    288      // [x:0..135][pad:136..159][gh:160..287]  = 9 * 32
#define NKC     9        // K chunks of 32 for wmma f16 16x16x32
#define GATES   512      // 4*H
#define EPSV    1e-5f
#define SLOPEV  0.01f

// GEMM tiling: block = 4 waves, 128 rows x 64 gate-cols; wave = 32 rows x 64 cols
#define GCOLS   64
#define GROWS   128

typedef __attribute__((ext_vector_type(16))) _Float16 v16h;
typedef __attribute__((ext_vector_type(8)))  _Float16 v8h;
typedef __attribute__((ext_vector_type(8)))  float    v8f;

// ---------------- workspace layout (bytes, all 256-aligned) ----------------
#define OFF_YF    ((size_t)0)                       // L*N*32 f16 = 25165824
#define OFF_YO    ((size_t)25165824)                // L*N*32 f16
#define OFF_SUMS  ((size_t)50331648)                // 128 f32 (sumF,sqF,sumO,sqO)
#define OFF_BNP   ((size_t)50332160)                // 128 f32 (scF,shF,scO,shO)
#define OFF_WCAT  ((size_t)50332672)                // 512*288 f16 = 294912
#define OFF_BIAS  ((size_t)50627584)                // 512 f32
#define OFF_ABUF  ((size_t)50629632)                // N*288 f16 = 9437184
#define OFF_GATES ((size_t)60066816)                // N*512 f32 = 33554432
#define OFF_GC    ((size_t)93621248)                // N*128 f32
#define OFF_H     ((size_t)102009856)               // N*128 f32
#define OFF_C     ((size_t)110398464)               // N*128 f32

__device__ __forceinline__ float sigmf(float x) { return 1.0f / (1.0f + __expf(-x)); }

// ---------------- 0: zero h, c, BN sums ----------------
__global__ void zero_state(float* __restrict__ h, float* __restrict__ c,
                           float* __restrict__ sums) {
    size_t i = (size_t)blockIdx.x * blockDim.x + threadIdx.x;
    const size_t HC = (size_t)NROWS * HDIM;
    if (i < HC)            h[i] = 0.0f;
    else if (i < 2 * HC)   c[i - HC] = 0.0f;
    else if (i < 2 * HC + 128) sums[i - 2 * HC] = 0.0f;
}

// ---------------- 1: fd/od pre-BN activations + BN statistics ----------------
__global__ __launch_bounds__(128)
void prep_mlp(const float* __restrict__ filt, const float* __restrict__ outp,
              const float* __restrict__ Wf, const float* __restrict__ bf,
              const float* __restrict__ Wo, const float* __restrict__ bo,
              _Float16* __restrict__ yf, _Float16* __restrict__ yo,
              float* __restrict__ sums) {
    __shared__ float s[128];   // sumF[32] sqF[32] sumO[32] sqO[32]
    if (threadIdx.x < 128) s[threadIdx.x] = 0.0f;
    __syncthreads();

    size_t r = (size_t)blockIdx.x * blockDim.x + threadIdx.x;  // row in [0, L*N)
    // ---- filter branch: 74 -> 32, leaky ----
    {
        float acc[SQ];
        #pragma unroll
        for (int ch = 0; ch < SQ; ++ch) acc[ch] = bf[ch];
        const float* fr = filt + r * D_FILT;
        for (int k = 0; k < D_FILT; ++k) {
            float v = fr[k];
            #pragma unroll
            for (int ch = 0; ch < SQ; ++ch) acc[ch] += v * Wf[ch * D_FILT + k];
        }
        #pragma unroll
        for (int ch = 0; ch < SQ; ++ch) {
            float y = acc[ch];
            y = (y > 0.0f) ? y : SLOPEV * y;
            yf[r * SQ + ch] = (_Float16)y;
            atomicAdd(&s[ch], y);
            atomicAdd(&s[32 + ch], y * y);
        }
    }
    // ---- output branch: 32 -> 32, leaky ----
    {
        float acc[SQ];
        #pragma unroll
        for (int ch = 0; ch < SQ; ++ch) acc[ch] = bo[ch];
        const float* orow = outp + r * D_OUT;
        for (int k = 0; k < D_OUT; ++k) {
            float v = orow[k];
            #pragma unroll
            for (int ch = 0; ch < SQ; ++ch) acc[ch] += v * Wo[ch * D_OUT + k];
        }
        #pragma unroll
        for (int ch = 0; ch < SQ; ++ch) {
            float y = acc[ch];
            y = (y > 0.0f) ? y : SLOPEV * y;
            yo[r * SQ + ch] = (_Float16)y;
            atomicAdd(&s[64 + ch], y);
            atomicAdd(&s[96 + ch], y * y);
        }
    }
    __syncthreads();
    if (threadIdx.x < 128) atomicAdd(&sums[threadIdx.x], s[threadIdx.x]);
}

// ---------------- 2: finalize BN scale/shift + fused bias ----------------
__global__ void finalize_params(const float* __restrict__ sums,
                                const float* __restrict__ g1, const float* __restrict__ beta1,
                                const float* __restrict__ g2, const float* __restrict__ beta2,
                                const float* __restrict__ bih, const float* __restrict__ bhh,
                                float* __restrict__ bnp, float* __restrict__ biascat) {
    int j = threadIdx.x;                       // block of 512
    biascat[j] = bih[j] + bhh[j];
    if (j < 64) {
        const float invLN = 1.0f / (float)((size_t)L_STEPS * NROWS);
        int  ch  = j & 31;
        bool isF = (j < 32);
        float sm = sums[isF ? ch : 64 + ch];
        float sq = sums[isF ? 32 + ch : 96 + ch];
        float mean = sm * invLN;
        float var  = sq * invLN - mean * mean;
        float sc = (isF ? g1[ch] : g2[ch]) * rsqrtf(var + EPSV);
        float sh = (isF ? beta1[ch] : beta2[ch]) - mean * sc;
        bnp[(isF ? 0 : 64) + ch]  = sc;
        bnp[(isF ? 32 : 96) + ch] = sh;
    }
}

// ---------------- 3: fused f16 weight [Wih | 0 | Whh] (512 x 288) ----------------
__global__ void build_wcat(const float* __restrict__ Wih, const float* __restrict__ Whh,
                           _Float16* __restrict__ Wcat) {
    int idx = blockIdx.x * blockDim.x + threadIdx.x;    // 512*288
    int j = idx / KPAD, k = idx % KPAD;
    float v;
    if (k < D_X)       v = Wih[j * D_X + k];
    else if (k < 160)  v = 0.0f;
    else               v = Whh[j * HDIM + (k - 160)];
    Wcat[(size_t)j * KPAD + k] = (_Float16)v;
}

// ---------------- 4: per-step A-matrix build + graph gather (gh into A, gc f32) ----------------
__global__ __launch_bounds__(128)
void build_A(const float* __restrict__ op, const float* __restrict__ attr,
             const _Float16* __restrict__ yf, const _Float16* __restrict__ yo,
             const float* __restrict__ bnp, const int* __restrict__ mapping,
             const float* __restrict__ h, const float* __restrict__ c,
             _Float16* __restrict__ Abuf, float* __restrict__ gc, int t) {
    int n = blockIdx.x * blockDim.x + threadIdx.x;      // row
    size_t base = (size_t)t * NROWS + n;
    _Float16* arow = Abuf + (size_t)n * KPAD;

    const float* oprow = op + base * D_OP;
    #pragma unroll
    for (int i = 0; i < D_OP; ++i) arow[i] = (_Float16)oprow[i];
    const float* atrow = attr + base * D_ATTR;
    for (int i = 0; i < D_ATTR; ++i) arow[D_OP + i] = (_Float16)atrow[i];

    const _Float16* yfr = yf + base * SQ;
    const _Float16* yor = yo + base * SQ;
    #pragma unroll
    for (int i = 0; i < SQ; ++i)
        arow[72 + i]  = (_Float16)((float)yfr[i] * bnp[i]      + bnp[32 + i]);
    #pragma unroll
    for (int i = 0; i < SQ; ++i)
        arow[104 + i] = (_Float16)((float)yor[i] * bnp[64 + i] + bnp[96 + i]);
    #pragma unroll
    for (int i = D_X; i < 160; ++i) arow[i] = (_Float16)0.0f;   // K padding

    int m0 = mapping[base * 2 + 0];
    int m1 = mapping[base * 2 + 1];
    float w0 = (m0 != 0) ? 0.5f : 0.0f;
    float w1 = (m1 != 0) ? 0.5f : 0.0f;
    int i0 = (m0 > 0) ? m0 - 1 : 0;
    int i1 = (m1 > 0) ? m1 - 1 : 0;
    const float* h0 = h + (size_t)i0 * HDIM; const float* h1 = h + (size_t)i1 * HDIM;
    const float* c0 = c + (size_t)i0 * HDIM; const float* c1 = c + (size_t)i1 * HDIM;
    float* gcr = gc + (size_t)n * HDIM;
    for (int j = 0; j < HDIM; ++j) {
        arow[160 + j] = (_Float16)(w0 * h0[j] + w1 * h1[j]);
        gcr[j]        = w0 * c0[j] + w1 * c1[j];
    }
}

// ---------------- 5: WMMA gate GEMM: [16384 x 288] @ [288 x 512] + bias ----------------
// Block: 4 waves, tile 128 rows x 64 cols. Wave: 32 rows x 64 cols
// (2 A fragments x 4 B tiles -> 8 f32 accumulators, 72 WMMAs/wave).
// B panel (64 cols x 288 K = 36 KB f16) staged to LDS via async-LDS loads.
__global__ __launch_bounds__(128)
void gemm_gates(const _Float16* __restrict__ Abuf, const _Float16* __restrict__ Wcat,
                const float* __restrict__ biascat, float* __restrict__ gates) {
    __shared__ __align__(32) _Float16 Bs[GCOLS * KPAD];   // 36864 B
    const int tid = threadIdx.x;
    const int cb  = blockIdx.y * GCOLS;                   // gate-column base

    // ---- async stage of 64 Wcat rows (= B columns) into LDS ----
    {
        const v8h* s = (const v8h*)(Wcat + (size_t)cb * KPAD);
        for (int i = tid; i < (GCOLS * KPAD) / 8; i += 128) {
            unsigned ldsoff = (unsigned)(size_t)((v8h*)Bs + i);   // LDS byte offset
            asm volatile("global_load_async_to_lds_b128 %0, %1, off"
                         :: "v"(ldsoff), "v"(s + i) : "memory");
        }
        asm volatile("s_wait_asynccnt 0x0" ::: "memory");
    }
    __syncthreads();

    const int lane = tid & 31, wave = tid >> 5;
    const int hi = lane >> 4, ln = lane & 15;
    const int m0 = blockIdx.x * GROWS + wave * 32;
    // A-fragment base (contiguous k order, same as B):
    //  lanes 0-15 hold K 0..15; lanes 16-31 hold K 16..31
    const _Float16* arow0 = Abuf + (size_t)(m0 + ln) * KPAD + hi * 16;
    const _Float16* arow1 = arow0 + (size_t)16 * KPAD;

    v8f acc[8];
    #pragma unroll
    for (int i = 0; i < 8; ++i) acc[i] = (v8f){0.f,0.f,0.f,0.f,0.f,0.f,0.f,0.f};

    #pragma unroll
    for (int kc = 0; kc < NKC; ++kc) {
        if (kc + 1 < NKC) {
            __builtin_prefetch(arow0 + (kc + 1) * 32, 0, 3);   // global_prefetch_b8
            __builtin_prefetch(arow1 + (kc + 1) * 32, 0, 3);
        }
        const v8h* ap0 = (const v8h*)(arow0 + kc * 32);
        const v8h* ap1 = (const v8h*)(arow1 + kc * 32);
        v8h a0lo = ap0[0], a0hi = ap0[1];
        v8h a1lo = ap1[0], a1hi = ap1[1];
        v16h a0, a1;
        #pragma unroll
        for (int i = 0; i < 8; ++i) {
            a0[i] = a0lo[i]; a0[8 + i] = a0hi[i];
            a1[i] = a1lo[i]; a1[8 + i] = a1hi[i];
        }
        #pragma unroll
        for (int cc = 0; cc < 4; ++cc) {
            // B 32x16: lane<16 -> col, K 0..15; lane>=16 -> col, K 16..31
            v16h b = *(const v16h*)(Bs + (size_t)(cc * 16 + ln) * KPAD
                                       + kc * 32 + hi * 16);
            acc[cc]     = __builtin_amdgcn_wmma_f32_16x16x32_f16(
                              false, a0, false, b, (short)0, acc[cc],     false, false);
            acc[4 + cc] = __builtin_amdgcn_wmma_f32_16x16x32_f16(
                              false, a1, false, b, (short)0, acc[4 + cc], false, false);
        }
    }
    // C/D layout: VGPR r -> M = r + 8*hi, N = lane&15
    #pragma unroll
    for (int cc = 0; cc < 4; ++cc) {
        int nn = cb + cc * 16 + ln;
        float bv = biascat[nn];
        #pragma unroll
        for (int r = 0; r < 8; ++r) {
            int m = m0 + r + 8 * hi;
            gates[(size_t)m * GATES + nn]        = acc[cc][r]     + bv;
            gates[(size_t)(m + 16) * GATES + nn] = acc[4 + cc][r] + bv;
        }
    }
}

// ---------------- 6: LSTM cell pointwise update ----------------
__global__ void lstm_cell(const float* __restrict__ gates, const float* __restrict__ gc,
                          float* __restrict__ h, float* __restrict__ c) {
    size_t idx = (size_t)blockIdx.x * blockDim.x + threadIdx.x;   // N*H
    size_t n = idx >> 7, j = idx & 127;
    const float* g = gates + n * GATES;
    float iv = sigmf(g[j]);
    float fv = sigmf(g[HDIM + j]);
    float gv = tanhf(g[2 * HDIM + j]);
    float ov = sigmf(g[3 * HDIM + j]);
    float cn = fv * gc[idx] + iv * gv;
    float hn = ov * tanhf(cn);
    c[idx] = cn;
    h[idx] = hn;
}

// ---------------- 7: output head: sigmoid((h@W1.T+b1)@W2.T+b2) ----------------
__global__ void head(const float* __restrict__ h,
                     const float* __restrict__ W1, const float* __restrict__ b1,
                     const float* __restrict__ W2, const float* __restrict__ b2,
                     float* __restrict__ out, int count) {
    int n = blockIdx.x * blockDim.x + threadIdx.x;
    if (n >= count) return;
    const float* hr = h + (size_t)n * HDIM;
    float z2 = b2[0];
    for (int k = 0; k < 20; ++k) {
        float z = b1[k];
        const float* wr = W1 + k * HDIM;
        for (int j = 0; j < HDIM; ++j) z += hr[j] * wr[j];
        z2 += z * W2[k];
    }
    out[n] = sigmf(z2);
}

// ---------------- launch ----------------
extern "C" void kernel_launch(void* const* d_in, const int* in_sizes, int n_in,
                              void* d_out, int out_size, void* d_ws, size_t ws_size,
                              hipStream_t stream) {
    const float* op    = (const float*)d_in[0];
    const float* attr  = (const float*)d_in[1];
    const float* filt  = (const float*)d_in[2];
    const float* outp  = (const float*)d_in[3];
    const int*   mapng = (const int*)  d_in[4];
    // d_in[5] = batch_size scalar; host-side out_size carries the same info
    const float* Wf    = (const float*)d_in[6];
    const float* bf    = (const float*)d_in[7];
    const float* Wo    = (const float*)d_in[8];
    const float* bo    = (const float*)d_in[9];
    const float* g1    = (const float*)d_in[10];
    const float* beta1 = (const float*)d_in[11];
    const float* g2    = (const float*)d_in[12];
    const float* beta2 = (const float*)d_in[13];
    const float* Wih   = (const float*)d_in[14];
    const float* Whh   = (const float*)d_in[15];
    const float* bih   = (const float*)d_in[16];
    const float* bhh   = (const float*)d_in[17];
    const float* W1    = (const float*)d_in[18];
    const float* b1    = (const float*)d_in[19];
    const float* W2    = (const float*)d_in[20];
    const float* b2    = (const float*)d_in[21];

    char* ws = (char*)d_ws;
    _Float16* yf      = (_Float16*)(ws + OFF_YF);
    _Float16* yo      = (_Float16*)(ws + OFF_YO);
    float*    sums    = (float*)   (ws + OFF_SUMS);
    float*    bnp     = (float*)   (ws + OFF_BNP);
    _Float16* Wcat    = (_Float16*)(ws + OFF_WCAT);
    float*    biascat = (float*)   (ws + OFF_BIAS);
    _Float16* Abuf    = (_Float16*)(ws + OFF_ABUF);
    float*    gates   = (float*)   (ws + OFF_GATES);
    float*    gc      = (float*)   (ws + OFF_GC);
    float*    hbuf    = (float*)   (ws + OFF_H);
    float*    cbuf    = (float*)   (ws + OFF_C);

    const size_t zn = (size_t)NROWS * HDIM * 2 + 128;
    zero_state<<<dim3((unsigned)((zn + 255) / 256)), 256, 0, stream>>>(hbuf, cbuf, sums);

    prep_mlp<<<dim3((L_STEPS * NROWS) / 128), 128, 0, stream>>>(
        filt, outp, Wf, bf, Wo, bo, yf, yo, sums);

    finalize_params<<<1, 512, 0, stream>>>(sums, g1, beta1, g2, beta2, bih, bhh,
                                           bnp, biascat);

    build_wcat<<<dim3((GATES * KPAD) / 256), 256, 0, stream>>>(Wih, Whh, Wcat);

    for (int t = L_STEPS - 1; t >= 0; --t) {
        build_A<<<dim3(NROWS / 128), 128, 0, stream>>>(
            op, attr, yf, yo, bnp, mapng, hbuf, cbuf, Abuf, gc, t);
        gemm_gates<<<dim3(NROWS / GROWS, GATES / GCOLS), 128, 0, stream>>>(
            Abuf, Wcat, biascat, gates);
        lstm_cell<<<dim3((NROWS * HDIM) / 256), 256, 0, stream>>>(
            gates, gc, hbuf, cbuf);
    }

    head<<<dim3((out_size + 127) / 128), 128, 0, stream>>>(
        hbuf, W1, b1, W2, b2, (float*)d_out, out_size);
}
